// SharedConsciousAgent_4836133175697
// MI455X (gfx1250) — matrix operand order repaired and checked by reading, not compiled, not run on hardware
//
#include <hip/hip_runtime.h>
#include <math.h>

// Problem constants (from reference)
#define BB 32
#define NA 64          // agents
#define DD 256
#define HDIM 8192
#define KMEM 16
#define NH 4
#define HDH 64         // head dim
#define BN (BB*NA)     // 2048

typedef __attribute__((ext_vector_type(16))) __bf16 v16bf;
typedef __attribute__((ext_vector_type(8)))  float  v8f;
typedef __attribute__((ext_vector_type(4)))  float  f4;   // native vector (NT-load capable)

static __device__ __forceinline__ float bf2f(unsigned short u) {
    unsigned int x = ((unsigned int)u) << 16; float f; __builtin_memcpy(&f, &x, 4); return f;
}
static __device__ __forceinline__ unsigned short f2bf(float f) {
    unsigned int x; __builtin_memcpy(&x, &f, 4);
    unsigned int r = x + 0x7fffu + ((x >> 16) & 1u);
    return (unsigned short)(r >> 16);
}
static __device__ __forceinline__ float wred(float v) {
    #pragma unroll
    for (int o = 16; o; o >>= 1) v += __shfl_xor(v, o);
    return v;
}
static __device__ __forceinline__ float sigmoidf_fast(float x) {
    return __builtin_amdgcn_rcpf(1.f + __expf(-x));
}
static __device__ __forceinline__ float tanhf_fast(float x) {
    // tanh(x) = 1 - 2/(exp(2x)+1); exact limits at +-inf
    return 1.f - 2.f * __builtin_amdgcn_rcpf(__expf(2.f * x) + 1.f);
}
static __device__ __forceinline__ float geluf(float x) { return 0.5f * x * (1.f + erff(x * 0.70710678118f)); }

// ---------------------------------------------------------------------------
// bf16 WMMA GEMM: C[M,N] = epi(A[M,K] @ W[K,N] + bias), W pre-transposed to [N][K].
// Block tile 64x64, 8 waves; each wave = 16x32 (2 WMMA acc sharing the A frag).
// Double-buffered LDS filled by GLOBAL_LOAD_ASYNC_TO_LDS_B128 (ASYNCcnt),
// overlapping the DMA of tile i+1 with the WMMAs of tile i.
// M%64==0, N%64==0, K%64==0 at every call site (K = 256/512/8192).
// EPI: 0=none 1=gelu 2=tanh 3=sigmoid 4=*(1+0.1*pers[row&63, col]) (N==256 only)
// ---------------------------------------------------------------------------
template<int EPI, bool WF, bool WB>
__global__ __launch_bounds__(256)
void k_gemm(const unsigned short* __restrict__ A,    // M x K row-major bf16
            const unsigned short* __restrict__ WT,   // N x K row-major bf16 (transposed W)
            const float* __restrict__ bias,          // N
            float* __restrict__ Cf,
            unsigned short* __restrict__ Cb,
            unsigned M, unsigned Nc, unsigned Kd,
            const float* __restrict__ pers) {
    __shared__ __align__(16) unsigned short As[2][64][40];   // pitch 80B (16B multiple)
    __shared__ __align__(16) unsigned short Bt[2][64][40];

    const unsigned tid  = threadIdx.x;
    const unsigned lane = tid & 31;
    const unsigned wave = tid >> 5;
    const unsigned wm = (wave >> 1) << 4;     // 0,16,32,48
    const unsigned wn = (wave & 1) << 5;      // 0,32
    const unsigned tileM = blockIdx.x << 6;
    const unsigned tileN = blockIdx.y << 6;
    const unsigned lr = tid >> 2;             // 0..63 (row of tile)
    const unsigned lk = (tid & 3) << 3;       // 0,8,16,24 (k offset, 16B aligned)

    const unsigned short* gA = A  + (tileM + lr) * Kd + lk;
    const unsigned short* gB = WT + (tileN + lr) * Kd + lk;
    const unsigned ldsA0 = (unsigned)(uintptr_t)&As[0][lr][lk];
    const unsigned ldsB0 = (unsigned)(uintptr_t)&Bt[0][lr][lk];
    const unsigned ldsA1 = (unsigned)(uintptr_t)&As[1][lr][lk];
    const unsigned ldsB1 = (unsigned)(uintptr_t)&Bt[1][lr][lk];

    const unsigned am = wm + (lane & 15);
    const unsigned kh = (lane >> 4) << 3;     // 0 / 8
    const unsigned bn = wn + (lane & 15);
    const unsigned kb = (lane >> 4) << 4;     // 0 / 16

    v8f acc0 = {}, acc1 = {};

    // async global->LDS: VDST = LDS byte address, VADDR = 64-bit global address
    auto issue = [&](unsigned ldsA, unsigned ldsB, unsigned k0) {
        unsigned long long a = (unsigned long long)(uintptr_t)(gA + k0);
        unsigned long long b = (unsigned long long)(uintptr_t)(gB + k0);
        asm volatile("global_load_async_to_lds_b128 %0, %1, off\n\t"
                     "global_load_async_to_lds_b128 %2, %3, off"
                     :: "v"(ldsA), "v"(a), "v"(ldsB), "v"(b) : "memory");
    };
    auto compute = [&](const unsigned short (&Asb)[64][40],
                       const unsigned short (&Btb)[64][40]) {
        union { uint4 u[2]; v16bf v; } af, b0, b1;
        // A frag 16x32: lane elems i = K kb+i, row = lane&15
        af.u[0] = *(const uint4*)&Asb[am][kb];
        af.u[1] = *(const uint4*)&Asb[am][kb + 8];
        // B frag 32x16 (from [n][k] tile): lane elems i = K kb+i, col = lane&15
        b0.u[0] = *(const uint4*)&Btb[bn][kb];
        b0.u[1] = *(const uint4*)&Btb[bn][kb + 8];
        b1.u[0] = *(const uint4*)&Btb[bn + 16][kb];
        b1.u[1] = *(const uint4*)&Btb[bn + 16][kb + 8];
        acc0 = __builtin_amdgcn_wmma_f32_16x16x32_bf16(false, af.v, false, b0.v, (short)0, acc0, false, false);
        acc1 = __builtin_amdgcn_wmma_f32_16x16x32_bf16(false, af.v, false, b1.v, (short)0, acc1, false, false);
    };

    const unsigned nIter = Kd >> 5;
    issue(ldsA0, ldsB0, 0);
    for (unsigned i = 0; i < nIter; i += 2) {
        asm volatile("s_wait_asynccnt 0x0" ::: "memory");
        __syncthreads();
        if (i + 1 < nIter) issue(ldsA1, ldsB1, (i + 1) << 5);
        compute(As[0], Bt[0]);
        if (i + 1 < nIter) {
            asm volatile("s_wait_asynccnt 0x0" ::: "memory");
            __syncthreads();
            if (i + 2 < nIter) issue(ldsA0, ldsB0, (i + 2) << 5);
            compute(As[1], Bt[1]);
        }
    }

    // C layout: VGPR r -> M = r + (lane>=16 ? 8:0) ; N = lane&15
    const unsigned gm0 = tileM + wm + ((lane >> 4) << 3);
    const unsigned gn0 = tileN + wn + (lane & 15);
    #pragma unroll
    for (int j = 0; j < 2; ++j) {
        const unsigned gn = gn0 + (j << 4);
        const float bc = bias[gn];
        #pragma unroll
        for (int r = 0; r < 8; ++r) {
            float v = (j ? acc1[r] : acc0[r]) + bc;
            if constexpr (EPI == 1) v = geluf(v);
            if constexpr (EPI == 2) v = tanhf_fast(v);
            if constexpr (EPI == 3) v = sigmoidf_fast(v);
            if constexpr (EPI == 4) v *= 1.f + 0.1f * pers[(((gm0 & 63) + r) << 8) + gn];
            const unsigned o = (gm0 + r) * Nc + gn;
            if constexpr (WF) Cf[o] = v;
            if constexpr (WB) Cb[o] = f2bf(v);
        }
    }
}

template<int EPI, bool WF, bool WB>
static inline void launch_gemm(const unsigned short* A, const unsigned short* WT,
                               const float* bias, float* Cf, unsigned short* Cb,
                               unsigned M, unsigned Nc, unsigned Kd,
                               const float* pers, hipStream_t s) {
    k_gemm<EPI, WF, WB><<<dim3(M >> 6, Nc >> 6), 256, 0, s>>>(A, WT, bias, Cf, Cb, M, Nc, Kd, pers);
}

// ---------------------------------------------------------------------------
// Conversions
// ---------------------------------------------------------------------------
// f32 (K x N) -> bf16 transposed (N x K). i indexes the input (coalesced reads).
__global__ void k_cvt_t(const float* __restrict__ s, unsigned short* __restrict__ d,
                        unsigned Kd, int lgN, unsigned total) {
    unsigned i = blockIdx.x * 256 + threadIdx.x;
    if (i >= total) return;
    unsigned n = i & ((1u << lgN) - 1u);
    unsigned k = i >> lgN;
    d[n * Kd + k] = f2bf(s[i]);
}
// f32 -> bf16 (vectorized, n % 4 == 0)
__global__ void k_cvt_bf16(const float* __restrict__ s, unsigned short* __restrict__ d, unsigned n4) {
    unsigned i = blockIdx.x * 256 + threadIdx.x;
    if (i >= n4) return;
    f4 v = ((const f4*)s)[i];
    ushort4 o; o.x = f2bf(v.x); o.y = f2bf(v.y); o.z = f2bf(v.z); o.w = f2bf(v.w);
    ((ushort4*)d)[i] = o;
}
__global__ void k_sub_bf16(const float* __restrict__ a, const float* __restrict__ b,
                           unsigned short* __restrict__ d, unsigned n4) {
    unsigned i = blockIdx.x * 256 + threadIdx.x;
    if (i >= n4) return;
    f4 x = ((const f4*)a)[i], y = ((const f4*)b)[i];
    ushort4 o; o.x = f2bf(x.x - y.x); o.y = f2bf(x.y - y.y); o.z = f2bf(x.z - y.z); o.w = f2bf(x.w - y.w);
    ((ushort4*)d)[i] = o;
}
__global__ void k_add_bf16(const float* __restrict__ a, const float* __restrict__ b,
                           unsigned short* __restrict__ d, unsigned n4) {
    unsigned i = blockIdx.x * 256 + threadIdx.x;
    if (i >= n4) return;
    f4 x = ((const f4*)a)[i], y = ((const f4*)b)[i];
    ushort4 o; o.x = f2bf(x.x + y.x); o.y = f2bf(x.y + y.y); o.z = f2bf(x.z + y.z); o.w = f2bf(x.w + y.w);
    ((ushort4*)d)[i] = o;
}

// ---------------------------------------------------------------------------
// Fused elementwise kernels (all indices 32-bit, shifts/masks only)
// ---------------------------------------------------------------------------
// incoming = sig*(1+sur); modulated = incoming*(1+0.1*(pers+0.3*chr))
__global__ void k_modulate(const float* __restrict__ sig, const float* __restrict__ sur,
                           const float* __restrict__ pers, const float* __restrict__ chr,
                           float* __restrict__ incF, float* __restrict__ modF, unsigned n) {
    unsigned i = blockIdx.x * 256 + threadIdx.x;
    if (i >= n) return;
    unsigned p = ((i >> 8) & 63) * 256 + (i & 255);
    float inc = sig[i] * (1.f + sur[i]);
    incF[i] = inc;
    modF[i] = inc * (1.f + 0.1f * (pers[p] + 0.3f * chr[p]));
}
__global__ void k_modulate2(const float* __restrict__ incF, const float* __restrict__ ns,
                            float* __restrict__ modF, unsigned n) {
    unsigned i = blockIdx.x * 256 + threadIdx.x;
    if (i < n) modF[i] = incF[i] + 0.1f * ns[i];
}
__global__ void k_concat_bf16(const float* __restrict__ x, const float* __restrict__ y,
                              unsigned short* __restrict__ d, unsigned n) {
    unsigned i = blockIdx.x * 256 + threadIdx.x;
    if (i >= n) return;
    unsigned row = i >> 9, j = i & 511;
    float v = (j < DD) ? x[(row << 8) + j] : y[(row << 8) + (j - DD)];
    d[i] = f2bf(v);
}

// LN over concat[x, y*yscale], width 512 -> bf16. wave per row, 8 rows/block.
__global__ __launch_bounds__(256)
void k_ln_concat(const float* __restrict__ x, const float* __restrict__ y,
                 const float* __restrict__ yscale,
                 const float* __restrict__ g, const float* __restrict__ b,
                 unsigned short* __restrict__ out, unsigned rows) {
    unsigned row = blockIdx.x * 8 + (threadIdx.x >> 5);
    if (row >= rows) return;
    unsigned lane = threadIdx.x & 31;
    float ys = yscale ? yscale[row] : 1.f;
    float vals[16], s = 0.f, s2 = 0.f;
    #pragma unroll
    for (int e = 0; e < 16; ++e) {
        unsigned j = e * 32 + lane;
        float v = (j < DD) ? x[(row << 8) + j] : y[(row << 8) + (j - DD)] * ys;
        vals[e] = v; s += v; s2 += v * v;
    }
    s = wred(s); s2 = wred(s2);
    float m = s * (1.f / 512.f);
    float rs = rsqrtf(s2 * (1.f / 512.f) - m * m + 1e-5f);
    #pragma unroll
    for (int e = 0; e < 16; ++e) {
        unsigned j = e * 32 + lane;
        out[(row << 9) + j] = f2bf((vals[e] - m) * rs * g[j] + b[j]);
    }
}

// LN(x+y), width 256 -> f32 + bf16
__global__ __launch_bounds__(256)
void k_add_ln(const float* __restrict__ x, const float* __restrict__ y,
              const float* __restrict__ g, const float* __restrict__ b,
              float* __restrict__ outF, unsigned short* __restrict__ outB, unsigned rows) {
    unsigned row = blockIdx.x * 8 + (threadIdx.x >> 5);
    if (row >= rows) return;
    unsigned lane = threadIdx.x & 31;
    float vals[8], s = 0.f, s2 = 0.f;
    #pragma unroll
    for (int e = 0; e < 8; ++e) {
        unsigned j = e * 32 + lane;
        float v = x[(row << 8) + j] + y[(row << 8) + j];
        vals[e] = v; s += v; s2 += v * v;
    }
    s = wred(s); s2 = wred(s2);
    float m = s * (1.f / 256.f);
    float rs = rsqrtf(s2 * (1.f / 256.f) - m * m + 1e-5f);
    #pragma unroll
    for (int e = 0; e < 8; ++e) {
        unsigned j = e * 32 + lane;
        float v = (vals[e] - m) * rs * g[j] + b[j];
        outF[(row << 8) + j] = v;
        outB[(row << 8) + j] = f2bf(v);
    }
}

// new_states = LN(gate*osc*dec + (1-gate*osc)*prev) -> f32 + bf16
__global__ __launch_bounds__(256)
void k_state_ln(const float* __restrict__ gate, const float* __restrict__ dec,
                const float* __restrict__ prev, const float* __restrict__ osc_rate,
                const int* __restrict__ step,
                const float* __restrict__ g, const float* __restrict__ b,
                float* __restrict__ outF, unsigned short* __restrict__ outB, unsigned rows) {
    unsigned row = blockIdx.x * 8 + (threadIdx.x >> 5);
    if (row >= rows) return;
    unsigned lane = threadIdx.x & 31;
    float ph = fmaxf(cosf((float)step[0] * 1.57079632679f), 0.f);
    float osc = sigmoidf_fast(osc_rate[row & 63]);
    float mask = osc + (1.f - osc) * ph;
    float vals[8], s = 0.f, s2 = 0.f;
    #pragma unroll
    for (int e = 0; e < 8; ++e) {
        unsigned j = e * 32 + lane;
        unsigned o = (row << 8) + j;
        float gm = gate[o] * mask;
        float v = gm * dec[o] + (1.f - gm) * prev[o];
        vals[e] = v; s += v; s2 += v * v;
    }
    s = wred(s); s2 = wred(s2);
    float m = s * (1.f / 256.f);
    float rs = rsqrtf(s2 * (1.f / 256.f) - m * m + 1e-5f);
    #pragma unroll
    for (int e = 0; e < 8; ++e) {
        unsigned j = e * 32 + lane;
        float v = (vals[e] - m) * rs * g[j] + b[j];
        outF[(row << 8) + j] = v;
        outB[(row << 8) + j] = f2bf(v);
    }
}

// mem_strength = sigmoid(cos(ep, q*key) + cos(cr, q*key)); block per row over HDIM
__global__ __launch_bounds__(256)
void k_cos_strength(const float* __restrict__ ep, const float* __restrict__ cr,
                    const unsigned short* __restrict__ qb, const float* __restrict__ key,
                    float* __restrict__ strength) {
    unsigned row = blockIdx.x;
    unsigned kbase = (row & 63) << 13;
    float pe = 0.f, pc = 0.f, pq = 0.f, ne = 0.f, nc = 0.f;
    for (unsigned j = threadIdx.x; j < HDIM; j += 256) {
        float q = bf2f(qb[(row << 13) + j]);
        float k = key[kbase + j];
        float e = ep[(row << 13) + j];
        float c = cr[(row << 13) + j];
        float qk = q * k;
        pe += e * qk; pc += c * qk; pq += q * q; ne += e * e; nc += c * c;
    }
    __shared__ float sh[8];
    float sums[5] = {pe, pc, pq, ne, nc};
    float res[5];
    #pragma unroll
    for (int i = 0; i < 5; ++i) {
        float v = wred(sums[i]);
        __syncthreads();
        if ((threadIdx.x & 31) == 0) sh[threadIdx.x >> 5] = v;
        __syncthreads();
        if (threadIdx.x == 0) {
            float t = 0.f;
            for (int w = 0; w < 8; ++w) t += sh[w];
            sh[0] = t;
        }
        __syncthreads();
        res[i] = sh[0];
        __syncthreads();
    }
    if (threadIdx.x == 0) {
        float nq = sqrtf(res[2]);
        float c1 = res[0] / (sqrtf(res[3]) * nq + 1e-8f);
        float c2 = res[1] / (sqrtf(res[4]) * nq + 1e-8f);
        strength[row] = sigmoidf_fast(c1 + c2);
    }
}

// 4-head, 16-key attention; block (64 threads) per row. kk/vv bf16; out bf16.
__global__ __launch_bounds__(64)
void k_attn(const float* __restrict__ q, const unsigned short* __restrict__ kk,
            const unsigned short* __restrict__ vv, unsigned short* __restrict__ outB) {
    const unsigned row = blockIdx.x, t = threadIdx.x;
    const unsigned h = t >> 4, k = t & 15;
    __shared__ float sw[64];
    const f4*      q4 = (const f4*)(q + (row << 8) + (h << 6));
    const ushort4* k4 = (const ushort4*)(kk + ((((row << 4) + k) << 8) + (h << 6)));
    float s = 0.f;
    #pragma unroll
    for (int d = 0; d < 16; ++d) {
        f4 qq = q4[d];
        ushort4 kv = k4[d];
        s += qq.x * bf2f(kv.x) + qq.y * bf2f(kv.y) + qq.z * bf2f(kv.z) + qq.w * bf2f(kv.w);
    }
    sw[t] = s * 0.125f;   // 1/sqrt(64)
    __syncthreads();
    float mx = -1e30f;
    #pragma unroll
    for (int j = 0; j < 16; ++j) mx = fmaxf(mx, sw[(h << 4) + j]);
    float e = __expf(sw[t] - mx);
    __syncthreads();
    sw[t] = e;
    __syncthreads();
    float den = 0.f;
    #pragma unroll
    for (int j = 0; j < 16; ++j) den += sw[(h << 4) + j];
    // thread t -> output dims t*4 .. t*4+3 (same head h)
    float a0 = 0.f, a1 = 0.f, a2 = 0.f, a3 = 0.f;
    #pragma unroll
    for (int j = 0; j < 16; ++j) {
        float w = sw[(h << 4) + j];
        ushort4 v4 = *(const ushort4*)(vv + (((row << 4) + j) << 8) + (t << 2));
        a0 += w * bf2f(v4.x); a1 += w * bf2f(v4.y); a2 += w * bf2f(v4.z); a3 += w * bf2f(v4.w);
    }
    float inv = __builtin_amdgcn_rcpf(den);
    ushort4 o4;
    o4.x = f2bf(a0 * inv); o4.y = f2bf(a1 * inv); o4.z = f2bf(a2 * inv); o4.w = f2bf(a3 * inv);
    *(ushort4*)(outB + (row << 8) + (t << 2)) = o4;
}

// confidence[row] = ns[row,:] . w + b   (wave per row)
__global__ __launch_bounds__(256)
void k_rowdot(const float* __restrict__ x, const float* __restrict__ w,
              const float* __restrict__ b, float* __restrict__ out, unsigned rows) {
    unsigned row = blockIdx.x * 8 + (threadIdx.x >> 5);
    if (row >= rows) return;
    unsigned lane = threadIdx.x & 31;
    float s = 0.f;
    #pragma unroll
    for (int e = 0; e < 8; ++e) { unsigned j = e * 32 + lane; s += x[(row << 8) + j] * w[j]; }
    s = wred(s);
    if (lane == 0) out[row] = s + b[0];
}

// softmax(conf*5) over agents per batch; block (NA threads) per batch
__global__ __launch_bounds__(NA)
void k_comp_softmax(const float* __restrict__ conf, float* __restrict__ w) {
    unsigned b = blockIdx.x, t = threadIdx.x;
    __shared__ float sx[NA];
    float x = conf[(b << 6) + t] * 5.f;
    sx[t] = x; __syncthreads();
    float mx = -1e30f;
    #pragma unroll
    for (int j = 0; j < NA; ++j) mx = fmaxf(mx, sx[j]);
    float e = __expf(x - mx);
    __syncthreads();
    sx[t] = e; __syncthreads();
    float den = 0.f;
    #pragma unroll
    for (int j = 0; j < NA; ++j) den += sx[j];
    w[(b << 6) + t] = e / den;
}

__global__ void k_scale_rows(const float* __restrict__ a, const float* __restrict__ w,
                             float* __restrict__ out, unsigned n) {
    unsigned i = blockIdx.x * 256 + threadIdx.x;
    if (i < n) out[i] = a[i] * w[i >> 8];
}

// mem_gate[row] = sigmoid(ns.wmg[:256] + q.wmg[256:] + b); block per row
__global__ __launch_bounds__(256)
void k_memgate(const float* __restrict__ ns, const unsigned short* __restrict__ qb,
               const float* __restrict__ wmg, const float* __restrict__ bmg,
               float* __restrict__ out) {
    unsigned row = blockIdx.x;
    float s = 0.f;
    for (unsigned j = threadIdx.x; j < DD; j += 256) s += ns[(row << 8) + j] * wmg[j];
    for (unsigned j = threadIdx.x; j < HDIM; j += 256) s += bf2f(qb[(row << 13) + j]) * wmg[DD + j];
    s = wred(s);
    __shared__ float sh[8];
    if ((threadIdx.x & 31) == 0) sh[threadIdx.x >> 5] = s;
    __syncthreads();
    if (threadIdx.x == 0) {
        float t = 0.f;
        for (int w = 0; w < 8; ++w) t += sh[w];
        out[row] = sigmoidf_fast(t + bmg[0]);
    }
}

// new_episodics = 0.99*ep + mg[row]*exp_hdc*key[n]   (f4, non-temporal stream)
__global__ void k_episodic(const float* __restrict__ ep, const float* __restrict__ exph,
                           const float* __restrict__ key, const float* __restrict__ mg,
                           float* __restrict__ out, unsigned n4) {
    unsigned i = blockIdx.x * 256 + threadIdx.x;
    if (i >= n4) return;
    unsigned row = i >> 11;                                   // HDIM/4 = 2048 per row
    unsigned kidx = ((row & 63) << 11) + (i & 2047);
    float g = mg[row];
    f4 e = __builtin_nontemporal_load(&((const f4*)ep)[i]);
    f4 x = __builtin_nontemporal_load(&((const f4*)exph)[i]);
    f4 k = __builtin_nontemporal_load(&((const f4*)key)[kidx]);
    f4 o;
    o.x = 0.99f * e.x + g * x.x * k.x;
    o.y = 0.99f * e.y + g * x.y * k.y;
    o.z = 0.99f * e.z + g * x.z * k.z;
    o.w = 0.99f * e.w + g * x.w * k.w;
    __builtin_nontemporal_store(o, &((f4*)out)[i]);
}

// new_memory = concat(buffer[:,1:,:], ns[:,None,:])  (f4, non-temporal stream)
__global__ void k_newmem(const float* __restrict__ buf, const float* __restrict__ ns,
                         float* __restrict__ out, unsigned n4) {
    unsigned i = blockIdx.x * 256 + threadIdx.x;
    if (i >= n4) return;
    unsigned row = i >> 10;                   // KMEM*DD/4 = 1024 per row
    unsigned rem = i & 1023;
    unsigned k = rem >> 6, c4 = rem & 63;     // DD/4 = 64
    f4 v = (k < KMEM - 1)
        ? __builtin_nontemporal_load(&((const f4*)buf)[(row << 10) + ((k + 1) << 6) + c4])
        : ((const f4*)ns)[(row << 6) + c4];
    __builtin_nontemporal_store(v, &((f4*)out)[i]);
}

// ---------------------------------------------------------------------------
extern "C" void kernel_launch(void* const* d_in, const int* in_sizes, int n_in,
                              void* d_out, int out_size, void* d_ws, size_t ws_size,
                              hipStream_t stream) {
    (void)in_sizes; (void)n_in; (void)out_size; (void)ws_size;
    // Tensor inputs (setup_inputs order)
    const float* in_sig  = (const float*)d_in[0];
    const float* in_act  = (const float*)d_in[1];
    const float* in_ep   = (const float*)d_in[2];
    const float* in_cr   = (const float*)d_in[3];
    const float* in_prev = (const float*)d_in[4];
    const float* in_pred = (const float*)d_in[5];
    const float* in_buf  = (const float*)d_in[6];
    const int*   in_step = (const int*)d_in[7];
    // params: jax pytree flatten -> dict keys alphabetical (nested too):
    // {"b","w"} -> b,w ; {"g","b"} -> b,g
    const float* act_b  = (const float*)d_in[8];
    const float* act_w  = (const float*)d_in[9];
    const float* atk_b  = (const float*)d_in[10];
    const float* atk_w  = (const float*)d_in[11];
    const float* atln_b = (const float*)d_in[12];
    const float* atln_g = (const float*)d_in[13];
    const float* ato_b  = (const float*)d_in[14];
    const float* ato_w  = (const float*)d_in[15];
    const float* atq_b  = (const float*)d_in[16];
    const float* atq_w  = (const float*)d_in[17];
    const float* atv_b  = (const float*)d_in[18];
    const float* atv_w  = (const float*)d_in[19];
    const float* chr_p  = (const float*)d_in[20];
    const float* comp_b = (const float*)d_in[21];
    const float* comp_w = (const float*)d_in[22];
    const float* d1_b   = (const float*)d_in[23];
    const float* d1_w   = (const float*)d_in[24];
    const float* d2_b   = (const float*)d_in[25];
    const float* d2_w   = (const float*)d_in[26];
    const float* dln_b  = (const float*)d_in[27];
    const float* dln_g  = (const float*)d_in[28];
    const float* fh_b   = (const float*)d_in[29];
    const float* fh_w   = (const float*)d_in[30];
    const float* hkey   = (const float*)d_in[31];
    const float* ln_b   = (const float*)d_in[32];
    const float* ln_g   = (const float*)d_in[33];
    const float* mg_b   = (const float*)d_in[34];
    const float* mg_w   = (const float*)d_in[35];
    const float* osc    = (const float*)d_in[36];
    const float* pl_b   = (const float*)d_in[37];
    const float* pl_w   = (const float*)d_in[38];
    const float* pln_b  = (const float*)d_in[39];
    const float* pln_g  = (const float*)d_in[40];
    const float* pers   = (const float*)d_in[41];
    const float* pr_b   = (const float*)d_in[42];
    const float* pr_w   = (const float*)d_in[43];
    const float* su_b   = (const float*)d_in[44];
    const float* su_w   = (const float*)d_in[45];
    const float* th_b   = (const float*)d_in[46];
    const float* th_w   = (const float*)d_in[47];
    const float* ug_b   = (const float*)d_in[48];
    const float* ug_w   = (const float*)d_in[49];

    // Output segments (float counts)
    float* out = (float*)d_out;
    float* o_actions = out;
    float* o_states  = out + (size_t)BN * DD;
    float* o_epis    = o_states + (size_t)BN * DD;
    float* o_exph    = o_epis + (size_t)BN * HDIM;
    float* o_mgate   = o_exph + (size_t)BN * HDIM;
    float* o_pred    = o_mgate + BN;
    float* o_newmem  = o_pred + (size_t)BN * DD;

    // Workspace bump allocator
    char* wsb = (char*)d_ws; size_t off = 0;
    auto alloc = [&](size_t bytes) -> void* {
        void* p = wsb + off; off = (off + bytes + 255) & ~(size_t)255; return p;
    };
    auto allocF = [&](size_t n) { return (float*)alloc(n * 4); };
    auto allocB = [&](size_t n) { return (unsigned short*)alloc(n * 2); };

    // bf16 transposed weights ([N][K])
    unsigned short* wSU = allocB(DD * DD);
    unsigned short* wPL = allocB(512 * DD);
    unsigned short* wTH = allocB((size_t)DD * HDIM);
    unsigned short* wFH = allocB((size_t)HDIM * DD);
    unsigned short* wD1 = allocB(512 * DD);
    unsigned short* wD2 = allocB(DD * DD);
    unsigned short* wUG = allocB(512 * DD);
    unsigned short* wAC = allocB(DD * DD);
    unsigned short* wQ  = allocB(DD * DD);
    unsigned short* wK  = allocB(DD * DD);
    unsigned short* wV  = allocB(DD * DD);
    unsigned short* wO  = allocB(DD * DD);
    unsigned short* wPR = allocB(DD * DD);
    // bf16 activations
    unsigned short* diffB  = allocB((size_t)BN * DD);
    unsigned short* ln512B = allocB((size_t)BN * 512);
    unsigned short* cat512B= allocB((size_t)BN * 512);
    unsigned short* percB  = allocB((size_t)BN * DD);
    unsigned short* h1B    = allocB((size_t)BN * DD);
    unsigned short* decB   = allocB((size_t)BN * DD);
    unsigned short* attnB  = allocB((size_t)BN * DD);
    unsigned short* ns0B   = allocB((size_t)BN * DD);
    unsigned short* ns1B   = allocB((size_t)BN * DD);
    unsigned short* hdcqB  = allocB((size_t)BN * HDIM);
    unsigned short* epcrB  = allocB((size_t)BN * HDIM);
    unsigned short* bufB   = allocB((size_t)BN * KMEM * DD);
    unsigned short* kkB    = allocB((size_t)BN * KMEM * DD);
    unsigned short* vvB    = allocB((size_t)BN * KMEM * DD);
    // f32 activations
    float* surF   = allocF((size_t)BN * DD);
    float* incF   = allocF((size_t)BN * DD);
    float* modF   = allocF((size_t)BN * DD);
    float* percF  = allocF((size_t)BN * DD);
    float* recF   = allocF((size_t)BN * DD);
    float* decF   = allocF((size_t)BN * DD);
    float* aprF   = allocF((size_t)BN * DD);
    float* dec2F  = allocF((size_t)BN * DD);
    float* qF     = allocF((size_t)BN * DD);
    float* gateF  = allocF((size_t)BN * DD);
    float* ns0F   = allocF((size_t)BN * DD);
    float* ns1F   = allocF((size_t)BN * DD);
    float* actF   = allocF((size_t)BN * DD);
    float* actFin = allocF((size_t)BN * DD);
    float* strF   = allocF(BN);
    float* confF  = allocF(BN);
    float* cwF    = allocF(BN);

    auto cvtT = [&](const float* s, unsigned short* d, unsigned Kd, int lgN) {
        unsigned total = Kd << lgN;
        k_cvt_t<<<dim3((total + 255) / 256), 256, 0, stream>>>(s, d, Kd, lgN, total);
    };
    const unsigned EW = 256;
    auto g4 = [&](unsigned n4) { return dim3((n4 + EW - 1) / EW); };

    // ---- weight convert+transpose (reused by 32..4096 blocks each) ----
    cvtT(su_w, wSU, DD, 8);
    cvtT(pl_w, wPL, 512, 8);
    cvtT(th_w, wTH, DD, 13);
    cvtT(fh_w, wFH, HDIM, 8);
    cvtT(d1_w, wD1, 512, 8);
    cvtT(d2_w, wD2, DD, 8);
    cvtT(ug_w, wUG, 512, 8);
    cvtT(act_w, wAC, DD, 8);
    cvtT(atq_w, wQ, DD, 8);
    cvtT(atk_w, wK, DD, 8);
    cvtT(atv_w, wV, DD, 8);
    cvtT(ato_w, wO, DD, 8);
    cvtT(pr_w, wPR, DD, 8);

    // ---- prologue ----
    const unsigned nBD = BN * DD, nBH = BN * HDIM, nBKD = BN * KMEM * DD;
    k_sub_bf16<<<g4(nBD / 4), EW, 0, stream>>>(in_sig, in_pred, diffB, nBD / 4);
    launch_gemm<3, true, false>(diffB, wSU, su_b, surF, nullptr, BN, DD, DD, nullptr, stream);
    k_modulate<<<g4(nBD), EW, 0, stream>>>(in_sig, surF, pers, chr_p, incF, modF, nBD);

    // round-invariant: K/V projections, ep+cr, from_hdc recall base
    k_cvt_bf16<<<g4(nBKD / 4), EW, 0, stream>>>(in_buf, bufB, nBKD / 4);
    launch_gemm<0, false, true>(bufB, wK, atk_b, nullptr, kkB, BN * KMEM, DD, DD, nullptr, stream);
    launch_gemm<0, false, true>(bufB, wV, atv_b, nullptr, vvB, BN * KMEM, DD, DD, nullptr, stream);
    k_add_bf16<<<g4(nBH / 4), EW, 0, stream>>>(in_ep, in_cr, epcrB, nBH / 4);
    launch_gemm<0, true, false>(epcrB, wFH, fh_b, recF, nullptr, BN, DD, HDIM, nullptr, stream);

    const unsigned LNB = BN / 8;
    const float* prevF = in_prev;
    for (int r = 0; r < 2; ++r) {
        const float* actions_cur = (r == 0) ? in_act : actF;
        float* nsF = (r == 0) ? ns0F : ns1F;
        unsigned short* nsB = (r == 0) ? ns0B : ns1B;

        k_ln_concat<<<LNB, 256, 0, stream>>>(modF, actions_cur, nullptr, pln_g, pln_b, ln512B, BN);
        launch_gemm<1, true, true>(ln512B, wPL, pl_b, percF, percB, BN, DD, 512, nullptr, stream);
        launch_gemm<2, false, true>(percB, wTH, th_b, nullptr, hdcqB, BN, HDIM, DD, nullptr, stream);
        k_cos_strength<<<BN, 256, 0, stream>>>(in_ep, in_cr, hdcqB, hkey, strF);
        k_ln_concat<<<LNB, 256, 0, stream>>>(percF, recF, strF, dln_g, dln_b, ln512B, BN);
        launch_gemm<1, false, true>(ln512B, wD1, d1_b, nullptr, h1B, BN, DD, 512, nullptr, stream);
        launch_gemm<4, true, true>(h1B, wD2, d2_b, decF, decB, BN, DD, DD, pers, stream);
        launch_gemm<0, true, false>(decB, wQ, atq_b, qF, nullptr, BN, DD, DD, nullptr, stream);
        k_attn<<<BN, 64, 0, stream>>>(qF, kkB, vvB, attnB);
        launch_gemm<0, true, false>(attnB, wO, ato_b, aprF, nullptr, BN, DD, DD, nullptr, stream);
        k_add_ln<<<LNB, 256, 0, stream>>>(decF, aprF, atln_g, atln_b, dec2F, decB, BN);
        k_concat_bf16<<<g4(BN * 512), EW, 0, stream>>>(prevF, dec2F, cat512B, BN * 512);
        launch_gemm<3, true, false>(cat512B, wUG, ug_b, gateF, nullptr, BN, DD, 512, nullptr, stream);
        k_state_ln<<<LNB, 256, 0, stream>>>(gateF, dec2F, prevF, osc, in_step, ln_g, ln_b, nsF, nsB, BN);
        if (r == 0) {
            k_modulate2<<<g4(nBD), EW, 0, stream>>>(incF, ns0F, modF, nBD);
            launch_gemm<0, true, false>(ns0B, wAC, act_b, actF, nullptr, BN, DD, DD, nullptr, stream);
            prevF = ns0F;
        }
    }

    // ---- epilogue ----
    launch_gemm<0, true, false>(ns1B, wAC, act_b, actFin, nullptr, BN, DD, DD, nullptr, stream);
    k_rowdot<<<LNB, 256, 0, stream>>>(ns1F, comp_w, comp_b, confF, BN);
    k_comp_softmax<<<BB, NA, 0, stream>>>(confF, cwF);
    k_scale_rows<<<g4(nBD), EW, 0, stream>>>(actFin, cwF, o_actions, nBD);
    (void)hipMemcpyAsync(o_states, ns1F, (size_t)nBD * 4, hipMemcpyDeviceToDevice, stream);
    launch_gemm<2, true, false>(ns1B, wTH, th_b, o_exph, nullptr, BN, HDIM, DD, nullptr, stream);
    k_memgate<<<BN, 256, 0, stream>>>(ns1F, hdcqB, mg_w, mg_b, o_mgate);
    k_episodic<<<g4(nBH / 4), EW, 0, stream>>>(in_ep, o_exph, hkey, o_mgate, o_epis, nBH / 4);
    launch_gemm<0, true, false>(ns1B, wPR, pr_b, o_pred, nullptr, BN, DD, DD, nullptr, stream);
    k_newmem<<<g4(nBKD / 4), EW, 0, stream>>>(in_buf, ns1F, o_newmem, nBKD / 4);
}
